// ConditionalDLFactorized16_74680891343535
// MI455X (gfx1250) — matrix operand order repaired and checked by reading, not compiled, not run on hardware
//
#include <hip/hip_runtime.h>

#define C_DIM 512
#define O_DIM 512
#define RD_DIM 64
#define NE_DIM 8
#define COMMIT_SCALE 0.1f
// (RD/2) * log(2*pi*TAU), TAU = 1.0
#define LOG_CONST 58.8120661251f

typedef __attribute__((ext_vector_type(16))) __bf16 v16bf;
typedef __attribute__((ext_vector_type(8)))  __bf16 v8bf;
typedef __attribute__((ext_vector_type(8)))  float  v8f;

// ---------------------------------------------------------------------------
// fp32 -> bf16 (round-to-nearest-even), grid-stride
// ---------------------------------------------------------------------------
__global__ void cvt_f32_to_bf16(const float* __restrict__ src,
                                unsigned short* __restrict__ dst, int n) {
  int stride = gridDim.x * blockDim.x;
  for (int i = blockIdx.x * blockDim.x + threadIdx.x; i < n; i += stride) {
    unsigned int u = __float_as_uint(src[i]);
    u += 0x7FFFu + ((u >> 16) & 1u);
    dst[i] = (unsigned short)(u >> 16);
  }
}

__global__ void zero_loss_kernel(float* loss) { *loss = 0.f; }

// ---------------------------------------------------------------------------
// WMMA fragment loaders (bf16, wave32)
// A 16x32: lane l holds row (l&15); K run  [(l>>4)*16 .. +15] (contiguous)
// B 32x16: lane l holds col (l&15); K run  [(l>>4)*16 .. +15] (contiguous)
// ---------------------------------------------------------------------------
__device__ __forceinline__ v16bf load_a_frag(const unsigned short* __restrict__ row,
                                             int c0, int lane) {
  int base = c0 + ((lane >> 4) << 4);
  v8bf lo = *(const v8bf*)(row + base);
  v8bf hi = *(const v8bf*)(row + base + 8);
  v16bf a;
#pragma unroll
  for (int i = 0; i < 8; ++i) { a[i] = lo[i]; a[i + 8] = hi[i]; }
  return a;
}

__device__ __forceinline__ v16bf load_b_frag_global(const unsigned short* __restrict__ colrow,
                                                    int c0, int lane) {
  int base = c0 + ((lane >> 4) << 4);
  return *(const v16bf*)(colrow + base);
}

__device__ __forceinline__ v16bf combine8(v8bf lo, v8bf hi) {
  v16bf v;
#pragma unroll
  for (int i = 0; i < 8; ++i) { v[i] = lo[i]; v[i + 8] = hi[i]; }
  return v;
}

// ---------------------------------------------------------------------------
// GMM responsibilities: k = x @ map_w^T + map_b (WMMA), then
// d2 -> logresp -> logsumexp -> resp; loss += -COMMIT * sum(denom)
// 128 threads = 4 waves; each wave owns a 16-token tile.
// ---------------------------------------------------------------------------
__global__ __launch_bounds__(128) void gmm_resp_kernel(
    const unsigned short* __restrict__ xb,    // [N, C] bf16
    const unsigned short* __restrict__ mwb,   // [RD, C] bf16
    const float* __restrict__ map_b,          // [RD]
    const float* __restrict__ cent,           // [NE, RD]
    const float* __restrict__ prior,          // [NE]
    float* __restrict__ resp,                 // [N, NE]
    float* __restrict__ loss) {
  __shared__ float kld[4][16][RD_DIM + 1];    // +1 pad: conflict-free row reads
  const int lane = threadIdx.x & 31;
  const int wave = threadIdx.x >> 5;
  const int mbase = (blockIdx.x * 4 + wave) * 16;
  const unsigned short* arow = xb + (size_t)(mbase + (lane & 15)) * C_DIM;

  v8f acc[4];
#pragma unroll
  for (int t = 0; t < 4; ++t) acc[t] = (v8f){};

  for (int c0 = 0; c0 < C_DIM; c0 += 32) {
    v16bf a = load_a_frag(arow, c0, lane);
#pragma unroll
    for (int t = 0; t < 4; ++t) {
      const unsigned short* brow = mwb + (size_t)(t * 16 + (lane & 15)) * C_DIM;
      v16bf b = load_b_frag_global(brow, c0, lane);
      acc[t] = __builtin_amdgcn_wmma_f32_16x16x32_bf16(
          false, a, false, b, (short)0, acc[t], false, false);
    }
  }

  const int rhalf = (lane >> 4) * 8;
#pragma unroll
  for (int t = 0; t < 4; ++t) {
    int col = t * 16 + (lane & 15);
    float mb = map_b[col];
#pragma unroll
    for (int v = 0; v < 8; ++v)
      kld[wave][rhalf + v][col] = acc[t][v] + mb;
  }
  __syncthreads();

  if (lane < 16) {
    float kr[RD_DIM];
    float nk = 0.f;
#pragma unroll
    for (int j = 0; j < RD_DIM; ++j) {
      kr[j] = kld[wave][lane][j];
      nk += kr[j] * kr[j];
    }
    float lr[NE_DIM], mx = -1e30f;
#pragma unroll
    for (int e = 0; e < NE_DIM; ++e) {
      float dot = 0.f, nc = 0.f;
#pragma unroll
      for (int j = 0; j < RD_DIM; ++j) {
        float cv = cent[e * RD_DIM + j];
        dot += kr[j] * cv;
        nc += cv * cv;
      }
      float d2 = nk + nc - 2.f * dot;
      float lrv = -0.5f * d2 - LOG_CONST + __logf(prior[e]);
      lr[e] = lrv;
      mx = fmaxf(mx, lrv);
    }
    float s = 0.f;
#pragma unroll
    for (int e = 0; e < NE_DIM; ++e) s += __expf(lr[e] - mx);
    float denom = mx + __logf(s);
    int token = mbase + lane;
#pragma unroll
    for (int e = 0; e < NE_DIM; ++e)
      resp[(size_t)token * NE_DIM + e] = __expf(lr[e] - denom);

    float sum = denom;
#pragma unroll
    for (int m = 8; m >= 1; m >>= 1) sum += __shfl_xor(sum, m, 16);
    if (lane == 0) atomicAdd(loss, -COMMIT_SCALE * sum);
  }
}

// ---------------------------------------------------------------------------
// Fused MoE GEMM:
//   y[n,o] = sum_e resp[n,e]*(x@W_e^T)[n,o] + (x@bias_w^T)[n,o] + bias_b[o]
// bias_w is expert #8 with implicit weight 1.
// Block = 4 waves sharing ONE 16-col output tile; each wave owns 32 tokens
// (two 16-row subtiles) -> 18 accumulators; each B fragment feeds 2 WMMAs.
// The 9 expert B tiles (9KB/K-step) are staged in LDS once per block via
// global_load_async_to_lds_b128 (ASYNCcnt), double-buffered so the async
// copy of step k+1 overlaps the 18 WMMAs of step k. Staging addresses are
// precomputed once per thread; stage() is just ptr+c0 / off+buf plus the
// async issue (wave-uniform slot count -> full EXEC, clean ASYNCcnt).
// LDS B row stride = 40 ushorts (80B): 16B-aligned chunks, rows hit
// distinct banks (20*r mod 64 distinct for r=0..15) -> conflict-free reads.
// ---------------------------------------------------------------------------
#define BROW 40                                // padded row stride (ushorts)
#define NSTAGE_CHUNKS (9 * 16 * 4)             // 576 x 16B chunks per K-step

__global__ __launch_bounds__(128) void moe_gemm_kernel(
    const unsigned short* __restrict__ xb,    // [N, C] bf16
    const unsigned short* __restrict__ pwb,   // [NE, O, C] bf16
    const unsigned short* __restrict__ bwb,   // [O, C] bf16
    const float* __restrict__ resp,           // [N, NE]
    const float* __restrict__ bias_b,         // [O]
    float* __restrict__ y) {                  // [N, O]
  __shared__ __attribute__((aligned(16))) unsigned short ldsb[2][9][16][BROW];

  const int tid  = threadIdx.x;
  const int lane = tid & 31;
  const int wave = tid >> 5;
  const int obase  = (blockIdx.x & 31) * 16;   // 32 output tiles
  const int mblock = (blockIdx.x >> 5) * 128;  // 128 tokens per block
  const int m0 = mblock + wave * 32;           // this wave's 2 subtiles
  const int r = lane & 15;

  const unsigned short* arow0 = xb + (size_t)(m0 + r) * C_DIM;
  const unsigned short* arow1 = arow0 + (size_t)16 * C_DIM;

  // ---- precompute this thread's staging slots (fixed across K-steps) ----
  const unsigned short* gbase[5];
  unsigned ldsoff[5];
  int nslots = 0;
  for (int q = tid; q < NSTAGE_CHUNKS; q += 128) {
    int e   = q >> 6;          // expert 0..8
    int row = (q >> 2) & 15;   // output row within tile
    int c16 = q & 3;           // 16B chunk within the 64B row
    gbase[nslots] =
        (e < NE_DIM ? pwb + ((size_t)e * O_DIM + obase + row) * C_DIM
                    : bwb + (size_t)(obase + row) * C_DIM) + c16 * 8;
    ldsoff[nslots] = (unsigned)(size_t)(&ldsb[0][e][row][c16 * 8]);
    ++nslots;                  // wave-uniform: 5 for waves 0-1, 4 for waves 2-3
  }
  const unsigned bufstride = (unsigned)sizeof(ldsb[0]);

  auto stage = [&](int c0, int buf) {
    unsigned badd = (unsigned)buf * bufstride;
    for (int i = 0; i < nslots; ++i) {
      asm volatile("global_load_async_to_lds_b128 %0, %1, off"
                   :: "v"(ldsoff[i] + badd), "v"(gbase[i] + c0)
                   : "memory");
    }
  };

  v8f acc[9][2];
#pragma unroll
  for (int e = 0; e < 9; ++e) {
    acc[e][0] = (v8f){};
    acc[e][1] = (v8f){};
  }

  stage(0, 0);

#pragma unroll 2
  for (int step = 0; step < C_DIM / 32; ++step) {
    const int c0 = step * 32;
    const int buf = step & 1;
    // wait for this wave's share of buf, then make all waves' shares visible
    asm volatile("s_wait_asynccnt 0x0" ::: "memory");
    __syncthreads();
    if (step + 1 < C_DIM / 32) stage(c0 + 32, buf ^ 1);

    v16bf a0 = load_a_frag(arow0, c0, lane);
    v16bf a1 = load_a_frag(arow1, c0, lane);
#pragma unroll
    for (int e = 0; e < 9; ++e) {
      const unsigned short* bp = &ldsb[buf][e][r][(lane >> 4) * 16];
      v16bf b = combine8(*(const v8bf*)bp, *(const v8bf*)(bp + 8));
      acc[e][0] = __builtin_amdgcn_wmma_f32_16x16x32_bf16(
          false, a0, false, b, (short)0, acc[e][0], false, false);
      acc[e][1] = __builtin_amdgcn_wmma_f32_16x16x32_bf16(
          false, a1, false, b, (short)0, acc[e][1], false, false);
    }
    // restaging this buffer next step is safe: the WMMAs above (and their DS
    // waits) precede the next iteration's barrier.
  }

  // Epilogue: VGPR v -> token row m0 + s*16 + v + 8*(lane>>4), col obase+(lane&15)
  const int rhalf = (lane >> 4) * 8;
  const float bb = bias_b[obase + r];
#pragma unroll
  for (int s = 0; s < 2; ++s) {
#pragma unroll
    for (int v = 0; v < 8; ++v) {
      int row = m0 + s * 16 + rhalf + v;
      const float* rp = resp + (size_t)row * NE_DIM;
      float out = acc[8][s][v] + bb;
#pragma unroll
      for (int e = 0; e < NE_DIM; ++e) out += rp[e] * acc[e][s][v];
      y[(size_t)row * O_DIM + (obase + r)] = out;
    }
  }
}

// ---------------------------------------------------------------------------
// Launch
// ---------------------------------------------------------------------------
extern "C" void kernel_launch(void* const* d_in, const int* in_sizes, int n_in,
                              void* d_out, int out_size, void* d_ws, size_t ws_size,
                              hipStream_t stream) {
  (void)n_in; (void)out_size; (void)ws_size;
  const float* x      = (const float*)d_in[0];
  // d_in[1] = key_feat: ignored by the reference forward
  const float* map_w  = (const float*)d_in[2];
  const float* map_b  = (const float*)d_in[3];
  const float* cent   = (const float*)d_in[4];
  const float* prior  = (const float*)d_in[5];
  const float* pw_w   = (const float*)d_in[6];
  const float* bias_w = (const float*)d_in[7];
  const float* bias_b = (const float*)d_in[8];

  const int N = in_sizes[0] / C_DIM;          // 16384 tokens

  // Workspace layout (bf16 copies + resp), ~21 MB total, 16B-aligned segments
  unsigned short* xb  = (unsigned short*)d_ws;                    // N*C
  unsigned short* pwb = xb  + (size_t)N * C_DIM;                  // NE*O*C
  unsigned short* bwb = pwb + (size_t)NE_DIM * O_DIM * C_DIM;     // O*C
  unsigned short* mwb = bwb + (size_t)O_DIM * C_DIM;              // RD*C
  float* resp = (float*)(mwb + (size_t)RD_DIM * C_DIM);           // N*NE fp32

  float* y    = (float*)d_out;
  float* loss = y + (size_t)N * O_DIM;

  cvt_f32_to_bf16<<<1024, 256, 0, stream>>>(x, xb, N * C_DIM);
  cvt_f32_to_bf16<<<1024, 256, 0, stream>>>(pw_w, pwb, NE_DIM * O_DIM * C_DIM);
  cvt_f32_to_bf16<<<256, 256, 0, stream>>>(bias_w, bwb, O_DIM * C_DIM);
  cvt_f32_to_bf16<<<32, 256, 0, stream>>>(map_w, mwb, RD_DIM * C_DIM);
  zero_loss_kernel<<<1, 1, 0, stream>>>(loss);

  gmm_resp_kernel<<<N / 64, 128, 0, stream>>>(xb, mwb, map_b, cent, prior, resp, loss);

  // 4096 blocks: (N/128) token blocks x 32 output tiles
  moe_gemm_kernel<<<(N / 128) * (O_DIM / 16), 128, 0, stream>>>(
      xb, pwb, bwb, resp, bias_b, y);
}
